// SparseStructureFlowModel_14843406975250
// MI455X (gfx1250) — compile-verified, matrix operand order checked by reading, not run on hardware
//
#include <hip/hip_runtime.h>

// ---------------------------------------------------------------------------
// SparseStructureFlowModel for MI455X (gfx1250, wave32, WMMA bf16->f32)
//
// S = 4101 tokens padded to SP = 4224 (multiple of 128) so every GEMM /
// attention tile is full; invalid self-attn keys are masked to -1e30.
// All big matmuls run through v_wmma_f32_16x16x32_bf16; attention is a
// flash-style streaming softmax with WMMA for both K^T*Q and P*V.
// bf16 operand tiles are staged with GLOBAL_LOAD_ASYNC_TO_LDS_B128 and
// double-buffered LDS (one barrier per K-step).
// ---------------------------------------------------------------------------

typedef __attribute__((ext_vector_type(16))) __bf16 v16bf;
typedef __attribute__((ext_vector_type(8)))  float  v8f;

#define SP   4224
#define SVAL 4101
#define CDIM 1024

__device__ __forceinline__ unsigned short f2bs(float f) {
    return __builtin_bit_cast(unsigned short, (__bf16)f);
}
__device__ __forceinline__ __bf16 bs2b(unsigned short u) {
    return __builtin_bit_cast(__bf16, u);
}

#define SWZF(x, imm) __builtin_bit_cast(float, __builtin_amdgcn_ds_swizzle(__builtin_bit_cast(int, (x)), (imm)))

__device__ __forceinline__ float wave_sum(float x) {
    x += SWZF(x, 0x041F);
    x += SWZF(x, 0x081F);
    x += SWZF(x, 0x101F);
    x += SWZF(x, 0x201F);
    x += SWZF(x, 0x401F);
    return x;
}
__device__ __forceinline__ float xor16_add(float x) { return x + SWZF(x, 0x401F); }
__device__ __forceinline__ float xor16_max(float x) { return fmaxf(x, SWZF(x, 0x401F)); }

__device__ __forceinline__ float bcastf(float x, int srcLane) {
    return __builtin_bit_cast(float, __builtin_amdgcn_ds_bpermute(srcLane << 2, __builtin_bit_cast(int, x)));
}

__device__ __forceinline__ v8f wmma_bf16(v16bf a, v16bf b, v8f c) {
    return __builtin_amdgcn_wmma_f32_16x16x32_bf16(false, a, false, b, (short)0, c, false, false);
}

// Async DMA: copy 16 bytes / lane from global into LDS, tracked by ASYNCcnt.
__device__ __forceinline__ void async_b128(void* ldsPtr, const void* gPtr) {
    unsigned lds = (unsigned)(size_t)ldsPtr;           // low 32 bits = LDS offset
    unsigned long long ga = (unsigned long long)gPtr;
    asm volatile("global_load_async_to_lds_b128 %0, %1, off"
                 :: "v"(lds), "v"(ga) : "memory");
}
__device__ __forceinline__ void async_wait0() {
    asm volatile("s_wait_asynccnt 0x0" ::: "memory");
}

// K-half mapping for 16-bit A/B fragments (wave32):
// lane<16: halves 0..7 -> K 0..7, halves 8..15 -> K 16..23
// lane>=16: +8.    kmap(h) = (h&7) + ((h&8)<<1) + khi,  khi = (lane>>4)*8
// ---------------------------------------------------------------------------

enum { EPI_F32 = 0, EPI_BF16_GELU = 1, EPI_RES_GATE = 2, EPI_RES = 3 };

// C[128x128 tile] = A[M x K](bf16) * B[K x N](f32->bf16) (+bias, epilogue)
// Double-buffered LDS; A staged by async-DMA, B converted through VGPRs.
template <int EPI>
__global__ __launch_bounds__(256) void gemm_kernel(
    const unsigned short* __restrict__ A, const float* __restrict__ B,
    const float* __restrict__ bias, const float* __restrict__ gate,
    float* __restrict__ outF, unsigned short* __restrict__ outB,
    int K, int Ntot)
{
    // stride 40 elems = 80B: 16-lane fragment reads conflict-free, rows 16B-aligned
    __shared__ __bf16 As[2][128][40];
    __shared__ __bf16 BsT[2][128][40];

    const int tid  = threadIdx.x;
    const int lane = tid & 31;
    const int wid  = tid >> 5;
    const int ln16 = lane & 15;
    const int khi  = (lane >> 4) << 3;  // 0 or 8
    const int wm   = (wid & 3) * 32;    // wave row offset in block tile
    const int wn   = (wid >> 2) * 64;   // wave col offset in block tile
    const int mBase = blockIdx.y * 128;
    const int nBase = blockIdx.x * 128;

    // A tile: 128 rows x 32 cols bf16 = 512 x 16B chunks, 2 per thread
    const int ar0 = tid >> 2, as0 = (tid & 3) << 3;             // chunk tid
    const int ar1 = (tid + 256) >> 2, as1 = ((tid + 256) & 3) << 3;
    // B tile: 32 rows x 128 cols f32, 16 per thread
    const int bk   = tid >> 3;           // 0..31
    const int bseg = (tid & 7) << 4;     // 0,16,..,112

    v8f acc[2][4] = {};
    float tv[16];

    auto issueA = [&](int k0, int buf) {
        async_b128(&As[buf][ar0][as0], A + (size_t)(mBase + ar0) * K + k0 + as0);
        async_b128(&As[buf][ar1][as1], A + (size_t)(mBase + ar1) * K + k0 + as1);
    };
    auto loadB = [&](int k0) {
        const float4* bs = reinterpret_cast<const float4*>(B + (size_t)(k0 + bk) * Ntot + nBase + bseg);
        float4 b0 = bs[0], b1 = bs[1], b2 = bs[2], b3 = bs[3];
        tv[0]=b0.x; tv[1]=b0.y; tv[2]=b0.z; tv[3]=b0.w;
        tv[4]=b1.x; tv[5]=b1.y; tv[6]=b1.z; tv[7]=b1.w;
        tv[8]=b2.x; tv[9]=b2.y; tv[10]=b2.z; tv[11]=b2.w;
        tv[12]=b3.x; tv[13]=b3.y; tv[14]=b3.z; tv[15]=b3.w;
    };
    auto storeB = [&](int buf) {
#pragma unroll
        for (int j = 0; j < 16; ++j) BsT[buf][bseg + j][bk] = (__bf16)tv[j];
    };

    // prologue: stage tile 0
    issueA(0, 0);
    loadB(0);
    storeB(0);
    async_wait0();
    __syncthreads();

    int buf = 0;
    for (int k0 = 0; k0 < K; k0 += 32) {
        const bool more = (k0 + 32) < K;
        if (more) {
            issueA(k0 + 32, buf ^ 1);   // DMA next A tile while computing
            loadB(k0 + 32);             // next B tile into registers
        }

        v16bf af[2], bf[4];
#pragma unroll
        for (int mi = 0; mi < 2; ++mi) {
            const int r = wm + mi * 16 + ln16;
#pragma unroll
            for (int h = 0; h < 16; ++h) af[mi][h] = As[buf][r][(h & 7) + ((h & 8) << 1) + khi];
        }
#pragma unroll
        for (int ni = 0; ni < 4; ++ni) {
            const int c = wn + ni * 16 + ln16;
#pragma unroll
            for (int h = 0; h < 16; ++h) bf[ni][h] = BsT[buf][c][(h & 7) + ((h & 8) << 1) + khi];
        }
#pragma unroll
        for (int mi = 0; mi < 2; ++mi)
#pragma unroll
            for (int ni = 0; ni < 4; ++ni)
                acc[mi][ni] = wmma_bf16(af[mi], bf[ni], acc[mi][ni]);

        if (more) {
            storeB(buf ^ 1);            // idle buffer: safe since last barrier
            async_wait0();
        }
        __syncthreads();
        buf ^= 1;
    }

    // epilogue: D layout row = khi + r, col = ln16 (+ tile offsets)
#pragma unroll
    for (int mi = 0; mi < 2; ++mi) {
#pragma unroll
        for (int ni = 0; ni < 4; ++ni) {
#pragma unroll
            for (int r = 0; r < 8; ++r) {
                const int gm = mBase + wm + mi * 16 + khi + r;
                const int gn = nBase + wn + ni * 16 + ln16;
                const size_t idx = (size_t)gm * Ntot + gn;
                float v = acc[mi][ni][r] + bias[gn];
                if (EPI == EPI_F32) {
                    outF[idx] = v;
                } else if (EPI == EPI_BF16_GELU) {
                    outB[idx] = f2bs(0.5f * v * (1.f + erff(v * 0.70710678118f)));
                } else if (EPI == EPI_RES_GATE) {
                    outF[idx] += gate[gn] * v;
                } else {  // EPI_RES
                    outF[idx] += v;
                }
            }
        }
    }
}

// ---------------------------------------------------------------------------
// Flash attention.  Q,K,V head-major bf16: [NH][rows][64]. O: [rows][1024] bf16.
// Scores computed as D[k,q] = Ktile(A) x Q^T(B): query lives on the lane axis
// so softmax stats are per-lane; score D-layout == A-fragment layout for P*V.
template <bool MASK>
__global__ __launch_bounds__(256) void flash_kernel(
    const unsigned short* __restrict__ Qg, const unsigned short* __restrict__ Kg,
    const unsigned short* __restrict__ Vg, unsigned short* __restrict__ Og,
    int SQ, int SK, int Svalid)
{
    __shared__ __bf16 Qs[128][72];
    __shared__ __bf16 Ks[32][72];     // 144B rows: 16B aligned for async B128
    __shared__ __bf16 VTs[64][40];

    const int tid = threadIdx.x, lane = tid & 31, wv = tid >> 5;
    const int ln16 = lane & 15;
    const int khi  = (lane >> 4) << 3;
    const int head = blockIdx.y;
    const int q0   = blockIdx.x * 128;

    const unsigned short* Qh = Qg + (size_t)head * SQ * 64;
    const unsigned short* Kh = Kg + (size_t)head * SK * 64;
    const unsigned short* Vh = Vg + (size_t)head * SK * 64;

    for (int i = tid; i < 128 * 64; i += 256) {
        const int r = i >> 6, d = i & 63;
        Qs[r][d] = bs2b(Qh[(size_t)(q0 + r) * 64 + d]);
    }
    __syncthreads();

    v16bf qf[2];
#pragma unroll
    for (int st = 0; st < 2; ++st)
#pragma unroll
        for (int h = 0; h < 16; ++h)
            qf[st][h] = Qs[wv * 16 + ln16][st * 32 + (h & 7) + ((h & 8) << 1) + khi];

    v8f o[4] = {};
    float m = -1e30f, l = 0.f;

    const int krow = tid >> 3;           // 0..31
    const int kseg = (tid & 7) << 3;     // 0,8,..,56 (elems; x2 = 16B chunks)

    for (int kc = 0; kc < SK; kc += 32) {
        __syncthreads();
        // K tile: async DMA (bf16 -> bf16, 16B per thread)
        async_b128(&Ks[krow][kseg], Kh + (size_t)(kc + krow) * 64 + kseg);
        // V tile: through VGPRs (transposed into VTs)
        for (int i = tid; i < 2048; i += 256) {
            const int kk = i >> 6, d = i & 63;
            VTs[d][kk] = bs2b(Vh[(size_t)(kc + kk) * 64 + d]);
        }
        async_wait0();
        __syncthreads();

        v8f s0 = {}, s1 = {};
#pragma unroll
        for (int st = 0; st < 2; ++st) {
            v16bf kf0, kf1;
#pragma unroll
            for (int h = 0; h < 16; ++h) {
                const int dd = st * 32 + (h & 7) + ((h & 8) << 1) + khi;
                kf0[h] = Ks[ln16][dd];
                kf1[h] = Ks[16 + ln16][dd];
            }
            s0 = wmma_bf16(kf0, qf[st], s0);
            s1 = wmma_bf16(kf1, qf[st], s1);
        }

        float p0[8], p1[8], mt = -1e30f;
#pragma unroll
        for (int r = 0; r < 8; ++r) {
            float a = s0[r] * 0.125f;   // 1/sqrt(64)
            float b = s1[r] * 0.125f;
            if (MASK) {
                const int kk = kc + khi + r;
                if (kk >= Svalid)      a = -1e30f;
                if (kk + 16 >= Svalid) b = -1e30f;
            }
            p0[r] = a; p1[r] = b;
            mt = fmaxf(mt, fmaxf(a, b));
        }
        mt = xor16_max(mt);                 // combine lane / lane^16 halves
        const float mn = fmaxf(m, mt);
        const float alpha = __expf(m - mn);
        float rs = 0.f;
#pragma unroll
        for (int r = 0; r < 8; ++r) {
            p0[r] = __expf(p0[r] - mn);
            p1[r] = __expf(p1[r] - mn);
            rs += p0[r] + p1[r];
        }
        rs = xor16_add(rs);
        l = l * alpha + rs;
        m = mn;

        float arow[8];
#pragma unroll
        for (int r = 0; r < 8; ++r) arow[r] = bcastf(alpha, khi + r);

        v16bf pf;                           // P in A-fragment layout (free!)
#pragma unroll
        for (int h = 0; h < 16; ++h) pf[h] = (__bf16)(h < 8 ? p0[h] : p1[h - 8]);

#pragma unroll
        for (int dt = 0; dt < 4; ++dt) {
            v16bf vf;
#pragma unroll
            for (int h = 0; h < 16; ++h)
                vf[h] = VTs[dt * 16 + ln16][(h & 7) + ((h & 8) << 1) + khi];
#pragma unroll
            for (int r = 0; r < 8; ++r) o[dt][r] *= arow[r];
            o[dt] = wmma_bf16(pf, vf, o[dt]);
        }
    }

    float linv[8];
#pragma unroll
    for (int r = 0; r < 8; ++r) linv[r] = 1.f / bcastf(l, khi + r);
#pragma unroll
    for (int dt = 0; dt < 4; ++dt)
#pragma unroll
        for (int r = 0; r < 8; ++r) {
            const int row = q0 + wv * 16 + khi + r;
            const int col = head * 64 + dt * 16 + ln16;
            Og[(size_t)row * CDIM + col] = f2bs(o[dt][r] * linv[r]);
        }
}

// ---------------------------------------------------------------------------
// LayerNorm: mode 0 -> y*(1+p0)+p1 (adaLN), mode 1 -> y*p0+p1 (weight/bias).
__global__ __launch_bounds__(256) void ln_kernel(
    const float* __restrict__ H, unsigned short* __restrict__ Ob,
    const float* __restrict__ p0, const float* __restrict__ p1, int mode)
{
    __shared__ float red[16];
    const int s = blockIdx.x, tid = threadIdx.x;
    const float* x = H + (size_t)s * CDIM;
    float xs[4], sm = 0.f, sq = 0.f;
#pragma unroll
    for (int j = 0; j < 4; ++j) { xs[j] = x[tid * 4 + j]; sm += xs[j]; sq += xs[j] * xs[j]; }
    sm = wave_sum(sm); sq = wave_sum(sq);
    const int wv = tid >> 5, lane = tid & 31;
    if (lane == 0) { red[wv] = sm; red[8 + wv] = sq; }
    __syncthreads();
    float tm = 0.f, tq = 0.f;
#pragma unroll
    for (int w = 0; w < 8; ++w) { tm += red[w]; tq += red[8 + w]; }
    const float mean = tm * (1.f / CDIM);
    const float var  = tq * (1.f / CDIM) - mean * mean;
    const float rstd = rsqrtf(fmaxf(var, 0.f) + 1e-6f);
#pragma unroll
    for (int j = 0; j < 4; ++j) {
        const int c = tid * 4 + j;
        const float y = (xs[j] - mean) * rstd;
        const float o = (mode == 0) ? y * (1.f + p0[c]) + p1[c] : y * p0[c] + p1[c];
        Ob[(size_t)s * CDIM + c] = f2bs(o);
    }
}

// RMS-norm q,k (+gamma), plain v; scatter to head-major bf16.  block=32.
__global__ void rmsqkv_kernel(
    const float* __restrict__ qkv, const float* __restrict__ qg, const float* __restrict__ kg,
    unsigned short* __restrict__ Qb, unsigned short* __restrict__ Kb, unsigned short* __restrict__ Vb)
{
    const int s = blockIdx.x, hh = blockIdx.y, lane = threadIdx.x;
    const size_t base = (size_t)s * 3072 + hh * 64;
    const float q0 = qkv[base + lane],        q1 = qkv[base + lane + 32];
    const float k0 = qkv[base + 1024 + lane], k1 = qkv[base + 1024 + lane + 32];
    const float v0 = qkv[base + 2048 + lane], v1 = qkv[base + 2048 + lane + 32];
    const float rq = rsqrtf(wave_sum(q0 * q0 + q1 * q1) * (1.f / 64.f) + 1e-6f);
    const float rk = rsqrtf(wave_sum(k0 * k0 + k1 * k1) * (1.f / 64.f) + 1e-6f);
    const size_t ob = ((size_t)hh * SP + s) * 64 + lane;
    Qb[ob]      = f2bs(q0 * rq * qg[hh * 64 + lane]);
    Qb[ob + 32] = f2bs(q1 * rq * qg[hh * 64 + lane + 32]);
    Kb[ob]      = f2bs(k0 * rk * kg[hh * 64 + lane]);
    Kb[ob + 32] = f2bs(k1 * rk * kg[hh * 64 + lane + 32]);
    Vb[ob]      = f2bs(v0);
    Vb[ob + 32] = f2bs(v1);
}

// fp32 [rows][nparts*1024] -> bf16 head-major [NH][rows][64] (select part).
__global__ void cvt_heads_kernel(const float* __restrict__ in, unsigned short* __restrict__ out,
                                 int rows, int part, int nparts)
{
    const int idx = blockIdx.x * 256 + threadIdx.x;
    if (idx >= rows * CDIM) return;
    const int s = idx >> 10, c = idx & 1023;
    const int hh = c >> 6, d = c & 63;
    out[((size_t)hh * rows + s) * 64 + d] = f2bs(in[(size_t)s * (nparts * CDIM) + part * CDIM + c]);
}

__global__ void cvt_bf16_kernel(const float* __restrict__ in, unsigned short* __restrict__ out, int n)
{
    const int i = blockIdx.x * 256 + threadIdx.x;
    if (i < n) out[i] = f2bs(in[i]);
}

// Token embedding + analytic 3D sin/cos pos-emb + special tokens + zero pad.
__global__ __launch_bounds__(256) void embed_kernel(
    const float* __restrict__ x, const float* __restrict__ in_w, const float* __restrict__ in_b,
    const float* __restrict__ pos_tok, const float* __restrict__ reg_tok, float* __restrict__ H)
{
    const int s = blockIdx.x, tid = threadIdx.x;
    float* hr = H + (size_t)s * CDIM;
#pragma unroll
    for (int j = 0; j < 4; ++j) {
        const int c = tid * 4 + j;
        float v;
        if (s == 0) {
            v = pos_tok[c];
        } else if (s < 5) {
            v = reg_tok[(s - 1) * CDIM + c];
        } else if (s < SVAL) {
            const int lidx = s - 5;
            v = in_b[c];
#pragma unroll
            for (int ic = 0; ic < 8; ++ic) v += x[ic * 4096 + lidx] * in_w[ic * CDIM + c];
            if (c < 1020) {
                const int axis = c / 340, r = c % 340, jj = r % 170;
                const int coord = (axis == 0) ? (lidx >> 8) : (axis == 1 ? ((lidx >> 4) & 15) : (lidx & 15));
                const float fr = __powf(10000.f, -(float)jj * (1.f / 170.f));
                const float a = (float)coord * fr;
                v += (r < 170) ? __sinf(a) : __cosf(a);
            }
        } else {
            v = 0.f;
        }
        hr[c] = v;
    }
}

// t -> freq embed -> fc1+silu -> fc2.  Single block.
__global__ __launch_bounds__(256) void time_kernel(
    const float* __restrict__ t, const float* __restrict__ w1, const float* __restrict__ b1,
    const float* __restrict__ w2, const float* __restrict__ b2, float* __restrict__ temb)
{
    __shared__ float tf[256];
    __shared__ float a1[1024];
    const int tid = threadIdx.x;
    const float tv = t[0];
    {
        const int j = tid & 127;
        const float fr = __expf(-logf(10000.f) * (float)j * (1.f / 128.f));
        const float a = tv * fr;
        tf[tid] = (tid < 128) ? __cosf(a) : __sinf(a);
    }
    __syncthreads();
#pragma unroll
    for (int j = 0; j < 4; ++j) {
        const int n = j * 256 + tid;
        float acc = b1[n];
        for (int k = 0; k < 256; ++k) acc += tf[k] * w1[k * CDIM + n];
        a1[n] = acc / (1.f + __expf(-acc));
    }
    __syncthreads();
#pragma unroll
    for (int j = 0; j < 4; ++j) {
        const int n = j * 256 + tid;
        float acc = b2[n];
        for (int k = 0; k < 1024; ++k) acc += a1[k] * w2[k * CDIM + n];
        temb[n] = acc;
    }
}

// mod[i] = silu(temb) @ ada_w[i] + ada_b[i], all 4 blocks.  grid = 96.
__global__ __launch_bounds__(256) void mod_kernel(
    const float* __restrict__ temb, const float* __restrict__ ada_w,
    const float* __restrict__ ada_b, float* __restrict__ modo)
{
    __shared__ float st[1024];
    const int tid = threadIdx.x;
#pragma unroll
    for (int j = 0; j < 4; ++j) {
        const int k = j * 256 + tid;
        const float v = temb[k];
        st[k] = v / (1.f + __expf(-v));
    }
    __syncthreads();
    const int gid = blockIdx.x * 256 + tid;
    const int i = gid / 6144, n = gid % 6144;
    float acc = ada_b[(size_t)i * 6144 + n];
    const float* W = ada_w + (size_t)i * 1024 * 6144 + n;
    for (int k = 0; k < 1024; ++k) acc += st[k] * W[(size_t)k * 6144];
    modo[gid] = acc;
}

// Final LN + out projection (C -> 8), transpose to [OC][L].  grid = 4096.
__global__ __launch_bounds__(256) void final_kernel(
    const float* __restrict__ H, const float* __restrict__ ow,
    const float* __restrict__ ob, float* __restrict__ out)
{
    __shared__ float y[1024];
    __shared__ float red[16];
    const int lidx = blockIdx.x, s = 5 + lidx, tid = threadIdx.x;
    const float* x = H + (size_t)s * CDIM;
    float xs[4], sm = 0.f, sq = 0.f;
#pragma unroll
    for (int j = 0; j < 4; ++j) { xs[j] = x[tid * 4 + j]; sm += xs[j]; sq += xs[j] * xs[j]; }
    sm = wave_sum(sm); sq = wave_sum(sq);
    const int wv = tid >> 5, lane = tid & 31;
    if (lane == 0) { red[wv] = sm; red[8 + wv] = sq; }
    __syncthreads();
    float tm = 0.f, tq = 0.f;
#pragma unroll
    for (int w = 0; w < 8; ++w) { tm += red[w]; tq += red[8 + w]; }
    const float mean = tm * (1.f / CDIM);
    const float rstd = rsqrtf(fmaxf(tq * (1.f / CDIM) - mean * mean, 0.f) + 1e-6f);
#pragma unroll
    for (int j = 0; j < 4; ++j) y[tid * 4 + j] = (xs[j] - mean) * rstd;
    __syncthreads();
    float acc = 0.f;  // wave wv handles output channel wv
    for (int c = lane; c < 1024; c += 32) acc += y[c] * ow[c * 8 + wv];
    acc = wave_sum(acc);
    if (lane == 0) out[(size_t)wv * 4096 + lidx] = acc + ob[wv];
}

// ---------------------------------------------------------------------------
extern "C" void kernel_launch(void* const* d_in, const int* in_sizes, int n_in,
                              void* d_out, int out_size, void* d_ws, size_t ws_size,
                              hipStream_t stream)
{
    (void)in_sizes; (void)n_in; (void)out_size; (void)ws_size;

    const float* x_in   = (const float*)d_in[0];
    const float* t_in   = (const float*)d_in[1];
    const float* cond   = (const float*)d_in[2];
    const float* t_w1   = (const float*)d_in[3];
    const float* t_b1   = (const float*)d_in[4];
    const float* t_w2   = (const float*)d_in[5];
    const float* t_b2   = (const float*)d_in[6];
    const float* in_w   = (const float*)d_in[7];
    const float* in_b   = (const float*)d_in[8];
    const float* pos_tk = (const float*)d_in[9];
    const float* reg_tk = (const float*)d_in[10];
    const float* ada_w  = (const float*)d_in[11];
    const float* ada_b  = (const float*)d_in[12];
    const float* qkv_w  = (const float*)d_in[13];
    const float* qkv_b  = (const float*)d_in[14];
    const float* q_g    = (const float*)d_in[15];
    const float* k_g    = (const float*)d_in[16];
    const float* aow    = (const float*)d_in[17];
    const float* aob_b  = (const float*)d_in[18];
    const float* n2_w   = (const float*)d_in[19];
    const float* n2_b   = (const float*)d_in[20];
    const float* cq_w   = (const float*)d_in[21];
    const float* cq_b   = (const float*)d_in[22];
    const float* ckv_w  = (const float*)d_in[23];
    const float* ckv_b  = (const float*)d_in[24];
    const float* co_w   = (const float*)d_in[25];
    const float* co_b   = (const float*)d_in[26];
    const float* m_w1   = (const float*)d_in[27];
    const float* m_b1   = (const float*)d_in[28];
    const float* m_w2   = (const float*)d_in[29];
    const float* m_b2   = (const float*)d_in[30];
    const float* out_w  = (const float*)d_in[31];
    const float* out_b  = (const float*)d_in[32];
    float* out = (float*)d_out;

    // workspace carve-up
    char* p = (char*)d_ws;
    auto alloc = [&](size_t bytes) -> void* {
        void* r = (void*)p;
        p += (bytes + 255) & ~(size_t)255;
        return r;
    };
    const size_t SPC = (size_t)SP * CDIM;
    float*          h     = (float*)alloc(SPC * 4);
    unsigned short* hnb   = (unsigned short*)alloc(SPC * 2);
    unsigned short* qb    = (unsigned short*)alloc(SPC * 2);
    unsigned short* kb    = (unsigned short*)alloc(SPC * 2);
    unsigned short* vb    = (unsigned short*)alloc(SPC * 2);
    unsigned short* attnb = (unsigned short*)alloc(SPC * 2);
    unsigned short* condb = (unsigned short*)alloc((size_t)1024 * 1024 * 2);
    unsigned short* mlpb  = (unsigned short*)alloc((size_t)SP * 4096 * 2);
    float*          temb  = (float*)alloc(1024 * 4);
    float*          modw  = (float*)alloc(4 * 6144 * 4);
    float*          big   = (float*)alloc((size_t)SP * 3072 * 4);  // reused scratch
    float* qkv_f = big;                    // [SP][3072]
    float* cq_f  = big;                    // [SP][1024]
    float* kv_f  = big + SPC;              // [1024][2048]

    const dim3 blk(256);

    embed_kernel<<<dim3(SP), blk, 0, stream>>>(x_in, in_w, in_b, pos_tk, reg_tk, h);
    cvt_bf16_kernel<<<dim3(4096), blk, 0, stream>>>(cond, condb, 1024 * 1024);
    time_kernel<<<dim3(1), blk, 0, stream>>>(t_in, t_w1, t_b1, t_w2, t_b2, temb);
    mod_kernel<<<dim3(96), blk, 0, stream>>>(temb, ada_w, ada_b, modw);

    for (int i = 0; i < 4; ++i) {
        const float* mod = modw + (size_t)i * 6144;
        // 1. hn = LN(h)*(1+sc1)+sh1
        ln_kernel<<<dim3(SP), blk, 0, stream>>>(h, hnb, mod + 1024, mod + 0, 0);
        // 2. qkv = hn @ qkv_w + b
        gemm_kernel<EPI_F32><<<dim3(24, SP / 128), blk, 0, stream>>>(
            hnb, qkv_w + (size_t)i * 1024 * 3072, qkv_b + (size_t)i * 3072,
            nullptr, qkv_f, nullptr, 1024, 3072);
        // 3. rms(q,k)*gamma, scatter head-major bf16
        rmsqkv_kernel<<<dim3(SP, 16), dim3(32), 0, stream>>>(
            qkv_f, q_g + (size_t)i * 1024, k_g + (size_t)i * 1024, qb, kb, vb);
        // 4. self flash attention (mask padded keys)
        flash_kernel<true><<<dim3(SP / 128, 16), blk, 0, stream>>>(qb, kb, vb, attnb, SP, SP, SVAL);
        // 5. h += g1 * (attn @ ow + ob)
        gemm_kernel<EPI_RES_GATE><<<dim3(8, SP / 128), blk, 0, stream>>>(
            attnb, aow + (size_t)i * 1024 * 1024, aob_b + (size_t)i * 1024,
            mod + 2048, h, nullptr, 1024, 1024);
        // 6. hn = LN(h; n2_w, n2_b)
        ln_kernel<<<dim3(SP), blk, 0, stream>>>(h, hnb, n2_w + (size_t)i * 1024, n2_b + (size_t)i * 1024, 1);
        // 7. q = hn @ cq_w + b
        gemm_kernel<EPI_F32><<<dim3(8, SP / 128), blk, 0, stream>>>(
            hnb, cq_w + (size_t)i * 1024 * 1024, cq_b + (size_t)i * 1024,
            nullptr, cq_f, nullptr, 1024, 1024);
        cvt_heads_kernel<<<dim3((int)(SPC / 256)), blk, 0, stream>>>(cq_f, qb, SP, 0, 1);
        // 8. kv = cond @ ckv_w + b
        gemm_kernel<EPI_F32><<<dim3(16, 8), blk, 0, stream>>>(
            condb, ckv_w + (size_t)i * 1024 * 2048, ckv_b + (size_t)i * 2048,
            nullptr, kv_f, nullptr, 1024, 2048);
        cvt_heads_kernel<<<dim3(4096), blk, 0, stream>>>(kv_f, kb, 1024, 0, 2);
        cvt_heads_kernel<<<dim3(4096), blk, 0, stream>>>(kv_f, vb, 1024, 1, 2);
        // 9. cross flash attention
        flash_kernel<false><<<dim3(SP / 128, 16), blk, 0, stream>>>(qb, kb, vb, attnb, SP, 1024, 1024);
        // 10. h += cross @ co_w + b
        gemm_kernel<EPI_RES><<<dim3(8, SP / 128), blk, 0, stream>>>(
            attnb, co_w + (size_t)i * 1024 * 1024, co_b + (size_t)i * 1024,
            nullptr, h, nullptr, 1024, 1024);
        // 11. hn = LN(h)*(1+sc2)+sh2
        ln_kernel<<<dim3(SP), blk, 0, stream>>>(h, hnb, mod + 4096, mod + 3072, 0);
        // 12. mid = gelu(hn @ w1 + b1)  (bf16 out)
        gemm_kernel<EPI_BF16_GELU><<<dim3(32, SP / 128), blk, 0, stream>>>(
            hnb, m_w1 + (size_t)i * 1024 * 4096, m_b1 + (size_t)i * 4096,
            nullptr, nullptr, mlpb, 1024, 4096);
        // 13. h += g2 * (mid @ w2 + b2)
        gemm_kernel<EPI_RES_GATE><<<dim3(8, SP / 128), blk, 0, stream>>>(
            mlpb, m_w2 + (size_t)i * 4096 * 1024, m_b2 + (size_t)i * 1024,
            mod + 5120, h, nullptr, 4096, 1024);
    }

    final_kernel<<<dim3(4096), blk, 0, stream>>>(h, out_w, out_b, out);
}
